// MultiHeadAttention_44263932953129
// MI455X (gfx1250) — compile-verified, matrix operand order checked by reading, not run on hardware
//
#include <hip/hip_runtime.h>

// ---- problem constants (from reference) ----
#define B_  2
#define S_  2048
#define D_  1024
#define H_  16
#define DK_ 64

typedef __attribute__((ext_vector_type(16))) __bf16 v16bf;
typedef __attribute__((ext_vector_type(8)))  float  v8f;

union FragB16 { v16bf v; unsigned int u[8]; };

__device__ __forceinline__ unsigned short f2bf(float f) {
  unsigned u = __float_as_uint(f);
  unsigned r = u + 0x7FFFu + ((u >> 16) & 1u);   // round-to-nearest-even
  return (unsigned short)(r >> 16);
}

// CDNA5 async global->LDS copy (16B per lane), tracked by ASYNCcnt.
__device__ __forceinline__ void async_ld_b128(unsigned lds_off, unsigned long long gaddr) {
  asm volatile("global_load_async_to_lds_b128 %0, %1, off"
               :: "v"(lds_off), "v"(gaddr) : "memory");
}
template<int N>
__device__ __forceinline__ void wait_asynccnt() {
  asm volatile("s_wait_asynccnt %0" :: "i"(N) : "memory");
}

// Load one 16x32 bf16 A/B fragment (CDNA5 16-bit layout) from a row-major
// LDS tile. row_ptr = &tile[row][0] with 'row' = base + (lane&15);
// halves r<4: K = 2r + 8*hl ; r>=4: K = 2r+8 + 8*hl  (hl = lane>=16).
__device__ __forceinline__ void ldfrag(FragB16& f, const unsigned short* row_ptr,
                                       int hl, int kk) {
  const unsigned short* p = row_ptr + kk;
#pragma unroll
  for (int r = 0; r < 8; ++r) {
    int kp = (r < 4) ? (2 * r + 8 * hl) : (2 * r + 8 + 8 * hl);
    f.u[r] = *(const unsigned int*)(p + kp);     // dword-aligned (kp even)
  }
}

// ---------------------------------------------------------------------------
// Prep: elementwise f32 -> bf16
// ---------------------------------------------------------------------------
__global__ __launch_bounds__(256)
void cvt_bf16(const float* __restrict__ in, unsigned short* __restrict__ out, int n) {
  int i = (blockIdx.x * 256 + threadIdx.x) * 4;
  if (i + 3 < n) {
    float4 v = *(const float4*)(in + i);
    out[i + 0] = f2bf(v.x); out[i + 1] = f2bf(v.y);
    out[i + 2] = f2bf(v.z); out[i + 3] = f2bf(v.w);
  }
}

// ---------------------------------------------------------------------------
// Prep: f32 [R][C] -> bf16 transposed [C][R]  (32x32 LDS tiles)
// ---------------------------------------------------------------------------
__global__ __launch_bounds__(256)
void cvt_transpose(const float* __restrict__ in, unsigned short* __restrict__ out,
                   int R, int C) {
  __shared__ float tile[32][33];
  const int bx = blockIdx.x * 32, by = blockIdx.y * 32;
  const int tx = threadIdx.x & 31, ty = threadIdx.x >> 5;   // 32 x 8
#pragma unroll
  for (int rr = 0; rr < 32; rr += 8)
    tile[ty + rr][tx] = in[(size_t)(by + ty + rr) * C + bx + tx];
  __syncthreads();
#pragma unroll
  for (int rr = 0; rr < 32; rr += 8)
    out[(size_t)(bx + ty + rr) * R + by + tx] = f2bf(tile[tx][ty + rr]);
}

// ---------------------------------------------------------------------------
// Async double-buffered GEMM:  C[M,N] = A[M,K] @ Wt[N,K]^T (+ bias)
// All-bf16 inputs, f32 WMMA accumulate. Block tile 128x256, K-step 32,
// 8 waves (2 along M x 4 along N), wave tile 64x64 (16 WMMA / K-step).
// Tiles stream via global_load_async_to_lds_b128 + ASYNCcnt.
// OBF: 0 = f32 row-major out, 1 = bf16 row-major out,
//      2 = bf16 per-head transposed out Vt[(b,h,dk), s]
// ---------------------------------------------------------------------------
template<int OBF>
__global__ __launch_bounds__(256)
void gemm_async(const unsigned short* __restrict__ Ab,   // [M][K] bf16
                const unsigned short* __restrict__ Wt,   // [N][K] bf16
                const float* __restrict__ bias,
                void* __restrict__ Cp, int M, int N, int K) {
  __shared__ __align__(16) unsigned short As[2][128][32];   // 2 x 8KB
  __shared__ __align__(16) unsigned short Bs[2][256][32];   // 2 x 16KB
  const int tid  = threadIdx.x;
  const int lane = tid & 31;
  const int w    = tid >> 5;
  const int wr   = w & 1;          // 2 waves along M (64 rows each)
  const int wc   = w >> 1;         // 4 waves along N (64 cols each)
  const int hl   = lane >> 4;
  const int l15  = lane & 15;
  const int m0   = blockIdx.y * 128;
  const int n0   = blockIdx.x * 256;
  const int nk   = K >> 5;

  auto issue = [&](int kstep, int buf) {
    const int kk = kstep * 32;
#pragma unroll
    for (int e = 0; e < 2; ++e) {                 // A tile: 128 rows x 64B
      int idx = e * 256 + tid;
      int row = idx >> 2, c8 = (idx & 3) * 8;
      const unsigned short* g = Ab + (size_t)(m0 + row) * K + kk + c8;
      async_ld_b128((unsigned)(uintptr_t)&As[buf][row][c8],
                    (unsigned long long)(uintptr_t)g);
    }
#pragma unroll
    for (int e = 0; e < 4; ++e) {                 // B tile: 256 rows x 64B
      int idx = e * 256 + tid;
      int row = idx >> 2, c8 = (idx & 3) * 8;
      const unsigned short* g = Wt + (size_t)(n0 + row) * K + kk + c8;
      async_ld_b128((unsigned)(uintptr_t)&Bs[buf][row][c8],
                    (unsigned long long)(uintptr_t)g);
    }
  };

  v8f acc[4][4];
#pragma unroll
  for (int i = 0; i < 4; ++i)
#pragma unroll
    for (int j = 0; j < 4; ++j)
      acc[i][j] = (v8f){0.f,0.f,0.f,0.f,0.f,0.f,0.f,0.f};

  issue(0, 0);
  if (nk > 1) issue(1, 1);

  for (int k = 0; k < nk; ++k) {
    if (k + 1 < nk) wait_asynccnt<6>();           // own slice of tile k landed
    else            wait_asynccnt<0>();
    __syncthreads();                              // whole tile visible

    const unsigned short (*Asb)[32] = As[k & 1];
    const unsigned short (*Bsb)[32] = Bs[k & 1];
    FragB16 a[4];
#pragma unroll
    for (int i = 0; i < 4; ++i)
      ldfrag(a[i], &Asb[wr * 64 + i * 16 + l15][0], hl, 0);
#pragma unroll
    for (int j = 0; j < 4; ++j) {
      FragB16 bf_;
      ldfrag(bf_, &Bsb[wc * 64 + j * 16 + l15][0], hl, 0);
#pragma unroll
      for (int i = 0; i < 4; ++i)
        acc[i][j] = __builtin_amdgcn_wmma_f32_16x16x32_bf16(
            false, a[i].v, false, bf_.v, (short)0, acc[i][j], false, false);
    }
    __syncthreads();                              // everyone done reading buf
    if (k + 2 < nk) issue(k + 2, k & 1);
  }

#pragma unroll
  for (int i = 0; i < 4; ++i)
#pragma unroll
    for (int j = 0; j < 4; ++j)
#pragma unroll
      for (int r = 0; r < 8; ++r) {
        int row = m0 + wr * 64 + i * 16 + r + 8 * hl;  // C layout: M = r + 8*hl
        int col = n0 + wc * 64 + j * 16 + l15;         //           N = lane&15
        float v = acc[i][j][r] + (bias ? bias[col] : 0.f);
        if (OBF == 0) {
          ((float*)Cp)[(size_t)row * N + col] = v;
        } else if (OBF == 1) {
          ((unsigned short*)Cp)[(size_t)row * N + col] = f2bf(v);
        } else {                                       // per-head transposed V
          int b = row >> 11, s = row & (S_ - 1);       // S_ = 2048
          int h = col >> 6, dk = col & (DK_ - 1);
          ((unsigned short*)Cp)[((size_t)(b * H_ + h) * DK_ + dk) * S_ + s] = f2bf(v);
        }
      }
}

// ---------------------------------------------------------------------------
// Phrasal gate, step 1: neighbor dot products over D.
// ---------------------------------------------------------------------------
__global__ __launch_bounds__(256)
void gate_dots(const float* __restrict__ inter, const float* __restrict__ key,
               float* __restrict__ rl, float* __restrict__ rr) {
  const int b = blockIdx.y, i = blockIdx.x, t = threadIdx.x;
  const float* ib = inter + (size_t)b * S_ * D_;
  const float* kb = key + (size_t)b * S_ * D_;
  float s1 = 0.f, s2 = 0.f;
  for (int d = t; d < D_; d += 256) {
    s1 += ib[(size_t)(i + 1) * D_ + d] * kb[(size_t)i * D_ + d];
    s2 += ib[(size_t)i * D_ + d] * kb[(size_t)(i + 1) * D_ + d];
  }
  __shared__ float red1[256], red2[256];
  red1[t] = s1; red2[t] = s2;
  __syncthreads();
  for (int off = 128; off > 0; off >>= 1) {
    if (t < off) { red1[t] += red1[t + off]; red2[t] += red2[t + off]; }
    __syncthreads();
  }
  if (t == 0) {
    rl[(size_t)b * (S_ - 1) + i] = red1[0];
    rr[(size_t)b * (S_ - 1) + i] = red2[0];
  }
}

// ---------------------------------------------------------------------------
// Phrasal gate, step 2: pair softmax + log-sum -> gate[b].
// ---------------------------------------------------------------------------
__global__ __launch_bounds__(256)
void gate_final(const float* __restrict__ rl, const float* __restrict__ rr,
                float* __restrict__ gate) {
  const int b = blockIdx.x, t = threadIdx.x;
  const float* rlb = rl + (size_t)b * (S_ - 1);
  const float* rrb = rr + (size_t)b * (S_ - 1);
  float acc = 0.f;
  for (int jj = t; jj < S_ - 3; jj += 256) {
    float a1 = rlb[jj],     b1 = rrb[jj + 1];
    float a2 = rlb[jj + 1], b2 = rrb[jj + 2];
    float p1 = 1.f / (1.f + __expf(a1 - b1));   // pr[jj][1]
    float p0 = 1.f / (1.f + __expf(b2 - a2));   // pr[jj+1][0]
    float Pi = sqrtf(p1 * p0 + 1e-9f);
    acc += __logf(Pi + 1e-8f);
  }
  __shared__ float red[256];
  red[t] = acc;
  __syncthreads();
  for (int off = 128; off > 0; off >>= 1) {
    if (t < off) red[t] += red[t + off];
    __syncthreads();
  }
  if (t == 0) gate[b] = __expf(red[0]);
}

// ---------------------------------------------------------------------------
// Causal flash attention, bf16 WMMA, per (b, h, 128 q-rows) workgroup.
// K/V chunks (64 keys) stream via async-to-LDS, double-buffered w/ ASYNCcnt.
// V is pre-transposed per head: Vt[(b,h,dk), s].
// ---------------------------------------------------------------------------
__global__ __launch_bounds__(256)
void flash_attn(const unsigned short* __restrict__ Qp,
                const unsigned short* __restrict__ Kp,
                const unsigned short* __restrict__ Vt,
                const float* __restrict__ gate,
                unsigned short* __restrict__ Ctx) {
  __shared__ __align__(16) unsigned short Qs[128][64];      // [q][dk]      16KB
  __shared__ __align__(16) unsigned short Ks[2][64][64];    // [key][dk]   2x8KB
  __shared__ __align__(16) unsigned short Vs[2][64][64];    // [dk][key]   2x8KB
  __shared__ __align__(16) unsigned short Ps[8][16][64];    // per-wave P   16KB

  const int tid = threadIdx.x, lane = tid & 31, w = tid >> 5;
  const int hl = lane >> 4, l15 = lane & 15;
  const int qbase = blockIdx.x * 128;
  const int bh = blockIdx.y, b = bh / H_, h = bh % H_;
  const unsigned short* Qb  = Qp + (size_t)b * S_ * D_ + (size_t)h * DK_;
  const unsigned short* Kb  = Kp + (size_t)b * S_ * D_ + (size_t)h * DK_;
  const unsigned short* Vtb = Vt + (size_t)(b * H_ + h) * DK_ * S_;
  unsigned short*      Ctxb = Ctx + (size_t)b * S_ * D_ + (size_t)h * DK_;

  auto issueKV = [&](int j, int buf) {
    const int k0 = j * 64;
#pragma unroll
    for (int e = 0; e < 2; ++e) {
      int idx = e * 256 + tid;
      int row = idx >> 3, c8 = (idx & 7) * 8;      // 64 rows x 128B
      const unsigned short* gk = Kb + (size_t)(k0 + row) * D_ + c8;
      async_ld_b128((unsigned)(uintptr_t)&Ks[buf][row][c8],
                    (unsigned long long)(uintptr_t)gk);
      const unsigned short* gv = Vtb + (size_t)row * S_ + k0 + c8;
      async_ld_b128((unsigned)(uintptr_t)&Vs[buf][row][c8],
                    (unsigned long long)(uintptr_t)gv);
    }
  };

  // stage Q tile (128x64) via async, then K/V chunks 0 and 1
#pragma unroll
  for (int e = 0; e < 4; ++e) {
    int idx = e * 256 + tid;
    int row = idx >> 3, c8 = (idx & 7) * 8;
    const unsigned short* gq = Qb + (size_t)(qbase + row) * D_ + c8;
    async_ld_b128((unsigned)(uintptr_t)&Qs[row][c8],
                  (unsigned long long)(uintptr_t)gq);
  }
  const int jmax = qbase / 64 + 1;                 // causal chunk bound
  issueKV(0, 0);
  if (jmax >= 1) issueKV(1, 1);

  const int wq0 = w * 16;
  float m_[8], l_[8];
#pragma unroll
  for (int r = 0; r < 8; ++r) { m_[r] = -1e30f; l_[r] = 0.f; }
  v8f O[4];
#pragma unroll
  for (int t = 0; t < 4; ++t) O[t] = (v8f){0.f,0.f,0.f,0.f,0.f,0.f,0.f,0.f};

  for (int j = 0; j <= jmax; ++j) {
    if (j < jmax) wait_asynccnt<4>();              // Q + chunk j landed (in-order)
    else          wait_asynccnt<0>();
    __syncthreads();
    const int k0 = j * 64;
    const unsigned short (*Ksb)[64] = Ks[j & 1];
    const unsigned short (*Vsb)[64] = Vs[j & 1];

    const bool active = (k0 <= qbase + wq0 + 15);  // wave-uniform, EXEC stays full
    if (active) {
      // ---- S = Q K^T ----
      v8f sv[4];
#pragma unroll
      for (int n = 0; n < 4; ++n) {
        v8f s = (v8f){0.f,0.f,0.f,0.f,0.f,0.f,0.f,0.f};
#pragma unroll
        for (int ks = 0; ks < 2; ++ks) {
          FragB16 aq, bk;
          ldfrag(aq, &Qs[wq0 + l15][0], hl, 32 * ks);
          ldfrag(bk, &Ksb[16 * n + l15][0], hl, 32 * ks);
          s = __builtin_amdgcn_wmma_f32_16x16x32_bf16(
              false, aq.v, false, bk.v, (short)0, s, false, false);
        }
        sv[n] = s;
      }
      // ---- scale + causal mask ----
#pragma unroll
      for (int n = 0; n < 4; ++n)
#pragma unroll
        for (int r = 0; r < 8; ++r) {
          int qrow = qbase + wq0 + r + 8 * hl;
          int key  = k0 + 16 * n + l15;
          float x = sv[n][r] * 0.125f;             // 1/sqrt(64)
          sv[n][r] = (key > qrow) ? -1e30f : x;
        }
      // ---- online softmax: row max / correction ----
      float corr[8];
#pragma unroll
      for (int r = 0; r < 8; ++r) {
        float mx = fmaxf(fmaxf(sv[0][r], sv[1][r]), fmaxf(sv[2][r], sv[3][r]));
        mx = fmaxf(mx, __shfl_xor(mx, 1, 32));
        mx = fmaxf(mx, __shfl_xor(mx, 2, 32));
        mx = fmaxf(mx, __shfl_xor(mx, 4, 32));
        mx = fmaxf(mx, __shfl_xor(mx, 8, 32));
        float mn = fmaxf(m_[r], mx);
        corr[r] = __expf(m_[r] - mn);
        m_[r] = mn;
      }
      // ---- P = exp(S - m), row sums, stage P for PV WMMA ----
#pragma unroll
      for (int r = 0; r < 8; ++r) {
        float rs = 0.f;
#pragma unroll
        for (int n = 0; n < 4; ++n) {
          float p = __expf(sv[n][r] - m_[r]);
          rs += p;
          Ps[w][r + 8 * hl][16 * n + l15] = f2bf(p);
        }
        rs += __shfl_xor(rs, 1, 32);
        rs += __shfl_xor(rs, 2, 32);
        rs += __shfl_xor(rs, 4, 32);
        rs += __shfl_xor(rs, 8, 32);
        l_[r] = l_[r] * corr[r] + rs;
      }
#pragma unroll
      for (int t = 0; t < 4; ++t)
#pragma unroll
        for (int r = 0; r < 8; ++r) O[t][r] *= corr[r];

      asm volatile("s_wait_dscnt 0" ::: "memory"); // P store -> P load (in-wave)

      // ---- O += P @ V ----
#pragma unroll
      for (int ks = 0; ks < 2; ++ks) {
        FragB16 ap;
        ldfrag(ap, &Ps[w][l15][0], hl, 32 * ks);
#pragma unroll
        for (int t = 0; t < 4; ++t) {
          FragB16 bv;
          ldfrag(bv, &Vsb[16 * t + l15][0], hl, 32 * ks);
          O[t] = __builtin_amdgcn_wmma_f32_16x16x32_bf16(
              false, ap.v, false, bv.v, (short)0, O[t], false, false);
        }
      }
    }
    __syncthreads();                               // all reads of buf done
    if (j + 2 <= jmax) issueKV(j + 2, j & 1);
  }

  // ---- epilogue: O * gate / l -> ctx (bf16) ----
  const float g = gate[b];
#pragma unroll
  for (int t = 0; t < 4; ++t)
#pragma unroll
    for (int r = 0; r < 8; ++r) {
      int qrow = qbase + wq0 + r + 8 * hl;
      int col  = 16 * t + l15;
      float v = O[t][r] * g / l_[r];
      Ctxb[(size_t)qrow * D_ + col] = f2bf(v);
    }
}

// ---------------------------------------------------------------------------
extern "C" void kernel_launch(void* const* d_in, const int* in_sizes, int n_in,
                              void* d_out, int out_size, void* d_ws, size_t ws_size,
                              hipStream_t stream) {
  const float* query = (const float*)d_in[0];
  const float* key   = (const float*)d_in[1];
  const float* value = (const float*)d_in[2];
  // d_in[3] = causal mask (bool) -- causality computed analytically, unused
  const float* Wq = (const float*)d_in[4];
  const float* bq = (const float*)d_in[5];
  const float* Wk = (const float*)d_in[6];
  const float* bk = (const float*)d_in[7];
  const float* Wv = (const float*)d_in[8];
  const float* bv = (const float*)d_in[9];
  const float* Wo = (const float*)d_in[10];
  const float* bo = (const float*)d_in[11];
  const float* Wp = (const float*)d_in[12];

  const int M = B_ * S_;                 // 4096
  const size_t msd = (size_t)M * D_;     // 4M elements
  const size_t dd  = (size_t)D_ * D_;    // 1M elements

  char* ws = (char*)d_ws;
  size_t off = 0;
  auto alloc = [&](size_t bytes) -> void* {
    void* p = ws + off;
    off = (off + bytes + 255) & ~(size_t)255;
    return p;
  };
  unsigned short* qb  = (unsigned short*)alloc(msd * 2);  // bf16 inputs
  unsigned short* kb  = (unsigned short*)alloc(msd * 2);
  unsigned short* vb  = (unsigned short*)alloc(msd * 2);
  unsigned short* Wqt = (unsigned short*)alloc(dd * 2);   // bf16 W^T
  unsigned short* Wkt = (unsigned short*)alloc(dd * 2);
  unsigned short* Wvt = (unsigned short*)alloc(dd * 2);
  unsigned short* Wpt = (unsigned short*)alloc(dd * 2);
  unsigned short* Wot = (unsigned short*)alloc(dd * 2);
  unsigned short* Qp  = (unsigned short*)alloc(msd * 2);  // projections
  unsigned short* Kp  = (unsigned short*)alloc(msd * 2);
  unsigned short* Vtp = (unsigned short*)alloc(msd * 2);  // per-head transposed V
  unsigned short* Ctx = (unsigned short*)alloc(msd * 2);
  float*          Inter = (float*)alloc(msd * 4);
  float*          rL    = (float*)alloc((size_t)B_ * (S_ - 1) * 4);
  float*          rR    = (float*)alloc((size_t)B_ * (S_ - 1) * 4);
  float*          gate  = (float*)alloc((size_t)B_ * 4);
  (void)ws_size; (void)in_sizes; (void)n_in; (void)out_size;

  // ---- prep: bf16 copies + transposed bf16 weights ----
  cvt_bf16<<<dim3(msd / 1024), 256, 0, stream>>>(query, qb, (int)msd);
  cvt_bf16<<<dim3(msd / 1024), 256, 0, stream>>>(key,   kb, (int)msd);
  cvt_bf16<<<dim3(msd / 1024), 256, 0, stream>>>(value, vb, (int)msd);
  dim3 tgrid(D_ / 32, D_ / 32);
  cvt_transpose<<<tgrid, 256, 0, stream>>>(Wq, Wqt, D_, D_);
  cvt_transpose<<<tgrid, 256, 0, stream>>>(Wk, Wkt, D_, D_);
  cvt_transpose<<<tgrid, 256, 0, stream>>>(Wv, Wvt, D_, D_);
  cvt_transpose<<<tgrid, 256, 0, stream>>>(Wp, Wpt, D_, D_);
  cvt_transpose<<<tgrid, 256, 0, stream>>>(Wo, Wot, D_, D_);

  // ---- projections (async WMMA GEMMs) ----
  dim3 ggrid(D_ / 256, M / 128);   // (4, 32)
  gemm_async<1><<<ggrid, 256, 0, stream>>>(qb, Wqt, bq, Qp,    M, D_, D_);
  gemm_async<1><<<ggrid, 256, 0, stream>>>(kb, Wkt, bk, Kp,    M, D_, D_);
  gemm_async<2><<<ggrid, 256, 0, stream>>>(vb, Wvt, bv, Vtp,   M, D_, D_);
  gemm_async<0><<<ggrid, 256, 0, stream>>>(qb, Wpt, nullptr, Inter, M, D_, D_);

  // ---- phrasal gate ----
  gate_dots <<<dim3(S_ - 1, B_), 256, 0, stream>>>(Inter, key, rL, rR);
  gate_final<<<dim3(B_),         256, 0, stream>>>(rL, rR, gate);

  // ---- causal flash attention ----
  flash_attn<<<dim3(S_ / 128, B_ * H_), 256, 0, stream>>>(Qp, Kp, Vtp, gate, Ctx);

  // ---- output projection (bf16 ctx -> f32 out) ----
  gemm_async<0><<<ggrid, 256, 0, stream>>>(Ctx, Wot, bo, d_out, M, D_, D_);
}
